// MultiHeadEntityOPTAttention_80668075754217
// MI455X (gfx1250) — matrix operand order repaired and checked by reading, not compiled, not run on hardware
//
#include <hip/hip_runtime.h>
#include <hip/hip_bf16.h>
#include <stdint.h>

// Problem constants (from reference): B,T,E,NH,NA,RH = 32,256,256,8,64,64
#define B_  32
#define T_  256
#define E_  256
#define NH_ 8
#define NA_ 64

typedef __attribute__((ext_vector_type(16))) __bf16 bf16x16;
typedef __attribute__((ext_vector_type(8)))  float  f32x8;

union BfBits { unsigned short u; __bf16 h; };

__device__ __forceinline__ __bf16 us_to_bf(unsigned short u) { BfBits b; b.u = u; return b.h; }

__device__ __forceinline__ unsigned short f2bf(float f) {
  unsigned int x = __float_as_uint(f);
  x += 0x7fffu + ((x >> 16) & 1u);           // round-to-nearest-even
  return (unsigned short)(x >> 16);
}

// ---------------------------------------------------------------------------
// WMMA fragment helpers (wave32; A and B share a contiguous k order, not the ISA table's)
// All loaders below read K-CONTIGUOUS runs per lane -> b128 vector loads.
// ---------------------------------------------------------------------------
// A (16x32 bf16, MxK): lane&15 = M; lanes 0-15 hold K 0-15,
// lanes 16-31 hold K 16-31.  src row-major [M][ld]: one 32B run/lane.
__device__ __forceinline__ bf16x16 load_a_frag(const unsigned short* __restrict__ p,
                                               int ld, int lane) {
  const int m  = lane & 15;
  const int kb = (lane >> 4) << 4;           // 0 or 16
  bf16x16 a;
#pragma unroll
  for (int i = 0; i < 16; ++i) {
    const int k = kb + i;
    a[i] = us_to_bf(p[(size_t)m * ld + k]);
  }
  return a;
}

// B (32x16 bf16, KxN) sourced TRANSPOSED: src row-major [N][ld] (K inner).
// lane&15 = N; lanes 0-15 hold K 0-15, lanes 16-31 K 16-31: one 32B run/lane.
__device__ __forceinline__ bf16x16 load_bT_frag(const unsigned short* __restrict__ p,
                                                int ld, int lane) {
  const int n  = lane & 15;
  const int kb = (lane >> 4) << 4;           // 0 or 16
  bf16x16 b;
#pragma unroll
  for (int i = 0; i < 16; ++i)
    b[i] = us_to_bf(p[(size_t)n * ld + kb + i]);
  return b;
}

__device__ __forceinline__ f32x8 wmma_bf16(bf16x16 a, bf16x16 b, f32x8 c) {
  return __builtin_amdgcn_wmma_f32_16x16x32_bf16(false, a, false, b,
                                                 (short)0, c, false, false);
}

__device__ __forceinline__ uint4 pack8_bf16(const f32x8& a) {
  union { unsigned short us[8]; uint4 v; } u;
#pragma unroll
  for (int r = 0; r < 8; ++r) u.us[r] = f2bf(a[r]);
  return u.v;
}

// ---------------------------------------------------------------------------
// Kernels
// ---------------------------------------------------------------------------
__global__ void k_cvt_bf16(const float* __restrict__ src,
                           unsigned short* __restrict__ dst, int n) {
  int i = blockIdx.x * blockDim.x + threadIdx.x;
  const int stride = gridDim.x * blockDim.x;
  for (; i < n; i += stride) dst[i] = f2bf(src[i]);
}

// W [K,N] f32  ->  WT [N,K] bf16   (LDS-tiled transpose, coalesced both sides)
__global__ void k_cvt_wT(const float* __restrict__ W,
                         unsigned short* __restrict__ WT) {
  constexpr int K = E_, N = NH_ * E_;
  __shared__ unsigned short tile[32][33];
  const int n0 = blockIdx.x * 32, k0 = blockIdx.y * 32;
  const int tx = threadIdx.x & 31, ty = threadIdx.x >> 5;   // 32 x 8
#pragma unroll
  for (int i = 0; i < 32; i += 8)
    tile[ty + i][tx] = f2bf(W[(size_t)(k0 + ty + i) * N + n0 + tx]);
  __syncthreads();
#pragma unroll
  for (int i = 0; i < 32; i += 8)
    WT[(size_t)(n0 + ty + i) * K + k0 + tx] = tile[tx][ty + i];
}

// C = A[M,K] @ WT[N,K]^T,  M=8192 K=256 N=2048, all bf16, f32 accum.
// 128 thr = 4 waves; 64x64 block tile; 32x32 per wave (2x2 WMMA tiles).
// TRANS_OUT=false: C[m,n] bf16 (scatter b16).  TRANS_OUT=true: write V
// transposed as Vt[b,h,e,tk] with contiguous b128 stores per lane.
template <bool TRANS_OUT>
__global__ void k_gemm_proj(const unsigned short* __restrict__ A,
                            const unsigned short* __restrict__ WT,
                            unsigned short* __restrict__ C) {
  constexpr int K = E_, N = NH_ * E_;
  __shared__ __align__(16) unsigned short As[64 * 32];
  __shared__ __align__(16) unsigned short Bs[64 * 32];
  const int tid  = threadIdx.x;
  const int lane = tid & 31, wid = tid >> 5;
  const int wm = (wid >> 1) * 32, wn = (wid & 1) * 32;
  const int mBase = blockIdx.y * 64, nBase = blockIdx.x * 64;
  const int lrow = tid >> 1;                 // 0..63
  const int lcol = (tid & 1) * 16;           // 0 or 16 halves (32B)
  f32x8 c00 = {}, c01 = {}, c10 = {}, c11 = {};
  for (int k0 = 0; k0 < K; k0 += 32) {
    const unsigned short* ag = &A[(size_t)(mBase + lrow) * K + k0 + lcol];
    const unsigned short* bg = &WT[(size_t)(nBase + lrow) * K + k0 + lcol];
    *(uint4*)&As[lrow * 32 + lcol]     = *(const uint4*)ag;
    *(uint4*)&As[lrow * 32 + lcol + 8] = *(const uint4*)(ag + 8);
    *(uint4*)&Bs[lrow * 32 + lcol]     = *(const uint4*)bg;
    *(uint4*)&Bs[lrow * 32 + lcol + 8] = *(const uint4*)(bg + 8);
    if (k0 + 32 < K) {                       // -> global_prefetch_b8
      __builtin_prefetch(ag + 32, 0, 1);
      __builtin_prefetch(bg + 32, 0, 1);
    }
    __syncthreads();
    bf16x16 a0 = load_a_frag(&As[wm * 32], 32, lane);
    bf16x16 a1 = load_a_frag(&As[(wm + 16) * 32], 32, lane);
    bf16x16 b0 = load_bT_frag(&Bs[wn * 32], 32, lane);
    bf16x16 b1 = load_bT_frag(&Bs[(wn + 16) * 32], 32, lane);
    c00 = wmma_bf16(a0, b0, c00);
    c01 = wmma_bf16(a0, b1, c01);
    c10 = wmma_bf16(a1, b0, c10);
    c11 = wmma_bf16(a1, b1, c11);
    __syncthreads();
  }
  const int mr = (lane >> 4) << 3, nc = lane & 15;
  auto store_tile = [&](const f32x8& acc, int m0, int n0) {
    if (TRANS_OUT) {
      const int n = n0 + nc;                 // n = h*E + e (tile never crosses head)
      const int b = m0 >> 8, h = n >> 8, e = n & (E_ - 1);
      unsigned short* dst =
          C + ((size_t)((b * NH_ + h) * E_ + e)) * T_ + (m0 & (T_ - 1)) + mr;
      *(uint4*)dst = pack8_bf16(acc);        // 8 contiguous tk -> b128 store
    } else {
#pragma unroll
      for (int r = 0; r < 8; ++r)
        C[(size_t)(m0 + mr + r) * N + n0 + nc] = f2bf(acc[r]);
    }
  };
  store_tile(c00, mBase + wm,      nBase + wn);
  store_tile(c01, mBase + wm,      nBase + wn + 16);
  store_tile(c10, mBase + wm + 16, nBase + wn);
  store_tile(c11, mBase + wm + 16, nBase + wn + 16);
}

// S[b,h,tq,tk] = (Q_bh @ K_bh^T)/16, masked -> -1e30.  32x32 per wave (2x2).
__global__ void k_scores(const unsigned short* __restrict__ Q,
                         const unsigned short* __restrict__ Km,
                         const uint8_t* __restrict__ mask,
                         float* __restrict__ S) {
  const int lane = threadIdx.x & 31, wid = threadIdx.x >> 5;
  const int w   = blockIdx.x * 4 + wid;      // < 16384
  const int tk0 = (w & 7) * 32;
  const int tq0 = ((w >> 3) & 7) * 32;
  const int h   = (w >> 6) & 7;
  const int b   = w >> 9;
  const int ld  = NH_ * E_;
  const unsigned short* qp = Q  + ((size_t)(b * T_ + tq0) * NH_ + h) * E_;
  const unsigned short* kp = Km + ((size_t)(b * T_ + tk0) * NH_ + h) * E_;
  f32x8 c00 = {}, c01 = {}, c10 = {}, c11 = {};
#pragma unroll
  for (int e0 = 0; e0 < E_; e0 += 32) {
    bf16x16 a0 = load_a_frag(qp + e0, ld, lane);
    bf16x16 a1 = load_a_frag(qp + (size_t)16 * ld + e0, ld, lane);
    bf16x16 b0 = load_bT_frag(kp + e0, ld, lane);
    bf16x16 b1 = load_bT_frag(kp + (size_t)16 * ld + e0, ld, lane);
    c00 = wmma_bf16(a0, b0, c00);
    c01 = wmma_bf16(a0, b1, c01);
    c10 = wmma_bf16(a1, b0, c10);
    c11 = wmma_bf16(a1, b1, c11);
  }
  const int mr = (lane >> 4) << 3, nc = lane & 15;
  auto store_s = [&](const f32x8& acc, int tq, int tk0_) {
    const int tk = tk0_ + nc;
#pragma unroll
    for (int r = 0; r < 8; ++r) {
      const int tqr = tq + mr + r;
      float v = acc[r] * 0.0625f;            // 1/sqrt(E)
      if (mask[(size_t)(b * T_ + tqr) * T_ + tk]) v = -1.0e30f;
      S[(size_t)((b * NH_ + h) * T_ + tqr) * T_ + tk] = v;
    }
  };
  store_s(c00, tq0,      tk0);
  store_s(c01, tq0,      tk0 + 16);
  store_s(c10, tq0 + 16, tk0);
  store_s(c11, tq0 + 16, tk0 + 16);
}

// sparsemax per row of 256 via Michelot iterative simplex projection.
// One wave32 per row; masked entries are -1e30; fully-masked row -> zeros.
__global__ void k_sparsemax(const float* __restrict__ S,
                            unsigned short* __restrict__ attnb) {
  const int lane = threadIdx.x & 31, wid = threadIdx.x >> 5;
  const int row  = blockIdx.x * 8 + wid;     // < B*NH*T = 65536
  const float* z = S + (size_t)row * T_;
  unsigned short* out = attnb + (size_t)row * T_;
  float v[8];
#pragma unroll
  for (int j = 0; j < 8; ++j) v[j] = z[lane + 32 * j];

  float mx = v[0];
#pragma unroll
  for (int j = 1; j < 8; ++j) mx = fmaxf(mx, v[j]);
  for (int o = 16; o; o >>= 1) mx = fmaxf(mx, __shfl_xor(mx, o, 32));
  if (mx < -1.0e29f) {                       // row fully masked -> attn = 0
#pragma unroll
    for (int j = 0; j < 8; ++j) out[lane + 32 * j] = 0;
    return;
  }
  float s = 0.f;
#pragma unroll
  for (int j = 0; j < 8; ++j) s += v[j];
  for (int o = 16; o; o >>= 1) s += __shfl_xor(s, o, 32);

  float cnt = (float)T_;
  float tau = (s - 1.0f) / cnt;
  for (int it = 0; it < 256; ++it) {         // support strictly shrinks
    float s2 = 0.f, c2 = 0.f;
#pragma unroll
    for (int j = 0; j < 8; ++j)
      if (v[j] > tau) { s2 += v[j]; c2 += 1.0f; }
    for (int o = 16; o; o >>= 1) {
      s2 += __shfl_xor(s2, o, 32);
      c2 += __shfl_xor(c2, o, 32);
    }
    const float ntau = (s2 - 1.0f) / c2;
    const bool done = (c2 == cnt);           // wave-uniform
    cnt = c2; tau = ntau;
    if (done) break;
  }
#pragma unroll
  for (int j = 0; j < 8; ++j)
    out[lane + 32 * j] = f2bf(fmaxf(v[j] - tau, 0.0f));
}

// O[b,tq,h,e] = attn[b,h,tq,:] @ V[b,:,h,e], V pre-transposed Vt[b,h,e,tk].
// 32x32 per wave (2x2 WMMA tiles); all fragments contiguous per lane.
__global__ void k_av(const unsigned short* __restrict__ At,
                     const unsigned short* __restrict__ Vt,
                     float* __restrict__ O) {
  const int lane = threadIdx.x & 31, wid = threadIdx.x >> 5;
  const int w   = blockIdx.x * 4 + wid;      // < 16384
  const int e0  = (w & 7) * 32;
  const int tq0 = ((w >> 3) & 7) * 32;
  const int h   = (w >> 6) & 7;
  const int b   = w >> 9;
  const unsigned short* ap = At + ((size_t)(b * NH_ + h) * T_ + tq0) * T_;
  const unsigned short* vp = Vt + ((size_t)(b * NH_ + h) * E_ + e0) * T_;
  f32x8 c00 = {}, c01 = {}, c10 = {}, c11 = {};
  for (int tk0 = 0; tk0 < T_; tk0 += 32) {
    bf16x16 a0 = load_a_frag(ap + tk0, T_, lane);
    bf16x16 a1 = load_a_frag(ap + (size_t)16 * T_ + tk0, T_, lane);
    bf16x16 b0 = load_bT_frag(vp + tk0, T_, lane);
    bf16x16 b1 = load_bT_frag(vp + (size_t)16 * T_ + tk0, T_, lane);
    c00 = wmma_bf16(a0, b0, c00);
    c01 = wmma_bf16(a0, b1, c01);
    c10 = wmma_bf16(a1, b0, c10);
    c11 = wmma_bf16(a1, b1, c11);
  }
  const int mr = (lane >> 4) << 3, nc = lane & 15;
  auto store_o = [&](const f32x8& acc, int tq, int e) {
#pragma unroll
    for (int r = 0; r < 8; ++r)
      O[(size_t)((b * T_ + tq + mr + r) * NH_ + h) * E_ + e + nc] = acc[r];
  };
  store_o(c00, tq0,      e0);
  store_o(c01, tq0,      e0 + 16);
  store_o(c10, tq0 + 16, e0);
  store_o(c11, tq0 + 16, e0 + 16);
}

// x_agg[b,a,e] = sum_t (!mask[b,a,t]) * x[b,t,e], a < NA
__global__ void k_xagg(const uint8_t* __restrict__ mask,
                       const float* __restrict__ x,
                       float* __restrict__ xa) {
  const int b = blockIdx.x / NA_, a = blockIdx.x % NA_;
  const int e = threadIdx.x;
  const uint8_t* mrow = mask + (size_t)(b * T_ + a) * T_;
  const float* xb = x + (size_t)b * T_ * E_;
  float acc = 0.f;
  for (int t = 0; t < T_; ++t)
    if (!mrow[t]) acc += xb[(size_t)t * E_ + e];
  xa[(size_t)blockIdx.x * E_ + e] = acc;
}

// sel[b,a,:] = softmax(x_agg[b,a,:] @ fc_w + fc_b) over NH
__global__ void k_select(const float* __restrict__ xa,
                         const float* __restrict__ fw,
                         const float* __restrict__ fb,
                         float* __restrict__ sel) {
  __shared__ float red[256];
  __shared__ float logit[NH_];
  const int row = blockIdx.x;                // b*NA + a
  const int e = threadIdx.x;
  const float xv = xa[(size_t)row * E_ + e];
  for (int h = 0; h < NH_; ++h) {
    red[e] = xv * fw[(size_t)e * NH_ + h];
    __syncthreads();
    for (int off = 128; off; off >>= 1) {
      if (e < off) red[e] += red[e + off];
      __syncthreads();
    }
    if (e == 0) logit[h] = red[0] + fb[h];
    __syncthreads();
  }
  if (e == 0) {
    float m = logit[0];
#pragma unroll
    for (int h = 1; h < NH_; ++h) m = fmaxf(m, logit[h]);
    float ex[NH_], s = 0.f;
#pragma unroll
    for (int h = 0; h < NH_; ++h) { ex[h] = expf(logit[h] - m); s += ex[h]; }
    const float inv = 1.0f / s;
#pragma unroll
    for (int h = 0; h < NH_; ++h) sel[(size_t)row * NH_ + h] = ex[h] * inv;
  }
}

// y[b,t,e] = t<NA ? sum_h sel[b,t,h]*O[b,t,h,e] : mean_h O[b,t,h,e]
__global__ void k_combine(const float* __restrict__ O,
                          const float* __restrict__ sel,
                          float* __restrict__ y) {
  const int bt = blockIdx.x;                 // b*T + t
  const int t  = bt % T_;
  const int e  = threadIdx.x;
  const float* orow = O + (size_t)bt * NH_ * E_ + e;
  float acc = 0.f;
  if (t < NA_) {
    const int b = bt / T_;
    const float* s = sel + (size_t)(b * NA_ + t) * NH_;
#pragma unroll
    for (int h = 0; h < NH_; ++h) acc += s[h] * orow[(size_t)h * E_];
  } else {
#pragma unroll
    for (int h = 0; h < NH_; ++h) acc += orow[(size_t)h * E_];
    acc *= (1.0f / NH_);
  }
  y[(size_t)bt * E_ + e] = acc;
}

// ---------------------------------------------------------------------------
extern "C" void kernel_launch(void* const* d_in, const int* in_sizes, int n_in,
                              void* d_out, int out_size, void* d_ws, size_t ws_size,
                              hipStream_t stream) {
  (void)in_sizes; (void)n_in; (void)out_size; (void)ws_size;
  const float*   x    = (const float*)d_in[0];
  /* d_in[1] = h, only used by reference for n_agents (=64, compiled in) */
  const uint8_t* mask = (const uint8_t*)d_in[2];   // JAX bool: 1 byte/elem
  const float*   wq   = (const float*)d_in[3];
  const float*   wk   = (const float*)d_in[4];
  const float*   wv   = (const float*)d_in[5];
  const float*   fsw  = (const float*)d_in[6];
  const float*   fsb  = (const float*)d_in[7];
  float*         out  = (float*)d_out;

  char* p = (char*)d_ws;
  auto take = [&](size_t bytes) -> char* {
    char* r = p; p += (bytes + 255) & ~(size_t)255; return r;
  };
  const size_t nX = (size_t)B_ * T_ * E_;          // 2,097,152
  const size_t nW = (size_t)E_ * NH_ * E_;         //   524,288
  const size_t nQ = (size_t)B_ * T_ * NH_ * E_;    // 16,777,216
  const size_t nS = (size_t)B_ * NH_ * T_ * T_;    // 16,777,216

  unsigned short* xb  = (unsigned short*)take(nX * 2);
  unsigned short* wqt = (unsigned short*)take(nW * 2);   // W^T bf16 [N,K]
  unsigned short* wkt = (unsigned short*)take(nW * 2);
  unsigned short* wvt = (unsigned short*)take(nW * 2);
  unsigned short* qb  = (unsigned short*)take(nQ * 2);   // [B,T,NH,E]
  unsigned short* kb  = (unsigned short*)take(nQ * 2);   // [B,T,NH,E]
  unsigned short* vt  = (unsigned short*)take(nQ * 2);   // [B,NH,E,T]
  float*          sc  = (float*)take(nS * 4);
  unsigned short* ab  = (unsigned short*)take(nS * 2);
  float*          ob  = (float*)take(nQ * 4);
  float*          xa  = (float*)take((size_t)B_ * NA_ * E_ * 4);
  float*          sl  = (float*)take((size_t)B_ * NA_ * NH_ * 4);

  // 1) bf16 conversions (x straight; weights transposed for k-contig frags)
  k_cvt_bf16<<<(int)((nX + 255) / 256), 256, 0, stream>>>(x, xb, (int)nX);
  dim3 gt(NH_ * E_ / 32, E_ / 32);                 // (64, 8)
  k_cvt_wT<<<gt, 256, 0, stream>>>(wq, wqt);
  k_cvt_wT<<<gt, 256, 0, stream>>>(wk, wkt);
  k_cvt_wT<<<gt, 256, 0, stream>>>(wv, wvt);

  // 2) Q/K/V projections: [8192,256]x[256,2048] via WMMA bf16 (2x2/wave)
  dim3 gp(NH_ * E_ / 64, B_ * T_ / 64);            // (32, 128)
  k_gemm_proj<false><<<gp, 128, 0, stream>>>(xb, wqt, qb);
  k_gemm_proj<false><<<gp, 128, 0, stream>>>(xb, wkt, kb);
  k_gemm_proj<true ><<<gp, 128, 0, stream>>>(xb, wvt, vt);  // -> Vt[b,h,e,tk]

  // 3) scores = QK^T/sqrt(E) + mask   (16384 wave-tiles of 32x32)
  k_scores<<<4096, 128, 0, stream>>>(qb, kb, mask, sc);

  // 4) sparsemax rows (wave32 per row)
  k_sparsemax<<<8192, 256, 0, stream>>>(sc, ab);

  // 5) out = attn @ V (via Vt)
  k_av<<<4096, 128, 0, stream>>>(ab, vt, ob);

  // 6) agent aggregation + head-select softmax
  k_xagg<<<B_ * NA_, 256, 0, stream>>>(mask, x, xa);
  k_select<<<B_ * NA_, 256, 0, stream>>>(xa, fsw, fsb, sl);

  // 7) final combine -> [B,T,E]
  k_combine<<<B_ * T_, 256, 0, stream>>>(ob, sl, out);
}
